// IGNNetDefaultModel_44650480009258
// MI455X (gfx1250) — matrix operand order repaired and checked by reading, not compiled, not run on hardware
//
#include <hip/hip_runtime.h>
#include <hip/hip_bf16.h>

typedef __attribute__((ext_vector_type(16))) __bf16    v16bf;
typedef __attribute__((ext_vector_type(8)))  float     v8f;
typedef __attribute__((ext_vector_type(4)))  float     v4f;
typedef __attribute__((ext_vector_type(4)))  unsigned  v4u;
typedef __attribute__((ext_vector_type(8)))  unsigned  v8u;

#define NNODES 256
#define NBATCH 256
#define BNROWS 65536L   // NBATCH * NNODES

__device__ __forceinline__ unsigned pack_bf16(float lo, float hi) {
    unsigned short a = __builtin_bit_cast(unsigned short, (__bf16)lo);
    unsigned short b = __builtin_bit_cast(unsigned short, (__bf16)hi);
    return (unsigned)a | ((unsigned)b << 16);
}

// ---------------------------------------------------------------------------
// Tiled bf16 WMMA GEMM:  C[z] = act( A[z](MxK) * B[z](KxNc) + bias )
// Block = 128 threads (4 waves, 2x2). Block tile 64x64; each wave owns a
// 32x32 quadrant = four 16x16 WMMA tiles. K-step 32.
// REQUIRES: M % 64 == 0, Nc % 64 == 0, K % 32 == 0, lda/ldb % 4 == 0.
// LDS holds packed bf16 pairs in fragment order:
//   Asp[row][kpair]  (A fragment = 2 aligned b128 reads)
//   Bsp[col][kpair]  (B fragment = 2 aligned b128 reads)
// ---------------------------------------------------------------------------
__global__ __launch_bounds__(128)
void wmma_gemm(const float* __restrict__ A, long lda, long sA,
               const float* __restrict__ B, long ldb, long sB,
               float* __restrict__ C, long ldc, long sC, long coff,
               int M, int K, int Nc,
               const float* __restrict__ bias, int relu)
{
    __shared__ __align__(16) unsigned Asp[64][20];  // 64 rows x 16 kpairs (+pad)
    __shared__ __align__(16) unsigned Bsp[64][20];  // 64 cols x 16 kpairs (+pad)

    const int tid  = threadIdx.x;
    const int lane = tid & 31;
    const int wave = tid >> 5;
    const int wm   = wave >> 1;      // wave row quadrant
    const int wn   = wave & 1;       // wave col quadrant
    const int half = lane >> 4;      // lane half selects K region
    const int l15  = lane & 15;

    const long z = blockIdx.z;
    const float* Ab = A + z * sA;
    const float* Bb = B + z * sB;
    float*       Cb = C + z * sC;

    const int tM = blockIdx.y * 64;
    const int tN = blockIdx.x * 64;

    // staging coordinates (compile-time unrolled, no guards)
    const int a_kp4 = tid & 3;        // 8-k group within row
    const int a_row = tid >> 2;       // rows 0..31 (+32 on second chunk)
    const int b_n4  = tid & 15;       // 4-col group
    const int b_kp  = tid >> 4;       // kpair 0..7 (+8 on second chunk)

    v8f acc[4] = {};

    for (int kt = 0; kt < K; kt += 32) {
        // ---- stage A: 64 rows x 32 k, each thread 2 chunks of 8 k ----
        #pragma unroll
        for (int i = 0; i < 2; ++i) {
            const int row = a_row + i * 32;
            const float* ap = Ab + (long)(tM + row) * lda + kt + a_kp4 * 8;
            v4f f0 = *(const v4f*)(ap);
            v4f f1 = *(const v4f*)(ap + 4);
            v4u p;
            p[0] = pack_bf16(f0[0], f0[1]);
            p[1] = pack_bf16(f0[2], f0[3]);
            p[2] = pack_bf16(f1[0], f1[1]);
            p[3] = pack_bf16(f1[2], f1[3]);
            *(v4u*)&Asp[row][a_kp4 * 4] = p;
        }
        // ---- stage B: 32 k x 64 n, each thread 2 chunks of (2k x 4n) ----
        #pragma unroll
        for (int i = 0; i < 2; ++i) {
            const int kp = b_kp + i * 8;
            const float* bp = Bb + (long)(kt + kp * 2) * ldb + tN + b_n4 * 4;
            v4f r0 = *(const v4f*)(bp);
            v4f r1 = *(const v4f*)(bp + ldb);
            #pragma unroll
            for (int j = 0; j < 4; ++j)
                Bsp[b_n4 * 4 + j][kp] = pack_bf16(r0[j], r1[j]);
        }
        __syncthreads();

        // ---- fragments: two aligned b128 LDS reads each ----
        v16bf afr[2], bfr[2];
        #pragma unroll
        for (int s = 0; s < 2; ++s) {
            const int ar = wm * 32 + s * 16 + l15;
            v8u a;
            v4u lo = *(const v4u*)&Asp[ar][half * 4];
            v4u hi = *(const v4u*)&Asp[ar][8 + half * 4];
            a[0] = lo[0]; a[1] = lo[1]; a[2] = lo[2]; a[3] = lo[3];
            a[4] = hi[0]; a[5] = hi[1]; a[6] = hi[2]; a[7] = hi[3];
            afr[s] = __builtin_bit_cast(v16bf, a);

            const int bn = wn * 32 + s * 16 + l15;
            v8u b;
            v4u blo = *(const v4u*)&Bsp[bn][half * 4];
            v4u bhi = *(const v4u*)&Bsp[bn][8 + half * 4];
            b[0] = blo[0]; b[1] = blo[1]; b[2] = blo[2]; b[3] = blo[3];
            b[4] = bhi[0]; b[5] = bhi[1]; b[6] = bhi[2]; b[7] = bhi[3];
            bfr[s] = __builtin_bit_cast(v16bf, b);
        }

        #pragma unroll
        for (int sa = 0; sa < 2; ++sa)
            #pragma unroll
            for (int sb = 0; sb < 2; ++sb)
                acc[sa * 2 + sb] = __builtin_amdgcn_wmma_f32_16x16x32_bf16(
                    false, afr[sa], false, bfr[sb], (short)0,
                    acc[sa * 2 + sb], false, false);

        __syncthreads();
    }

    // ---- epilogue: C VGPR i -> row half*8+i, col l15 ----
    #pragma unroll
    for (int sa = 0; sa < 2; ++sa) {
        const int grow = tM + wm * 32 + sa * 16 + half * 8;
        #pragma unroll
        for (int sb = 0; sb < 2; ++sb) {
            const int gcol = tN + wn * 32 + sb * 16 + l15;
            const float bv = bias ? bias[gcol] : 0.0f;
            #pragma unroll
            for (int i = 0; i < 8; ++i) {
                float v = acc[sa * 2 + sb][i] + bv;
                if (relu) v = fmaxf(v, 0.0f);
                Cb[(long)(grow + i) * ldc + coff + gcol] = v;
            }
        }
    }
}

// ---------------------------------------------------------------------------
// skip_1 = x(BN,1) @ W_fst(1,64)  -> outer product, no GEMM needed
// ---------------------------------------------------------------------------
__global__ void outer64_kernel(const float* __restrict__ x,
                               const float* __restrict__ w,
                               float* __restrict__ S1)
{
    long idx = (long)blockIdx.x * blockDim.x + threadIdx.x;   // BN*64 total
    long r = idx >> 6;
    int  c = (int)(idx & 63);
    S1[idx] = x[r] * w[c];
}

// ---------------------------------------------------------------------------
// Per-sample BatchNorm over node axis: one block per sample, thread=channel.
// ---------------------------------------------------------------------------
__global__ void bn_kernel(const float* __restrict__ in, int ldin,
                          float* __restrict__ out, int ldout,
                          const float* __restrict__ g,
                          const float* __restrict__ b, int C)
{
    const int bi = blockIdx.x;
    const int c  = threadIdx.x;
    if (c >= C) return;
    const float* base = in + (long)bi * NNODES * ldin + c;

    float s = 0.0f;
    for (int n = 0; n < NNODES; ++n) s += base[(long)n * ldin];
    const float m = s * (1.0f / NNODES);

    float v = 0.0f;
    for (int n = 0; n < NNODES; ++n) {
        float d = base[(long)n * ldin] - m;
        v += d * d;
    }
    v *= (1.0f / NNODES);

    const float scale = rsqrtf(v + 1e-5f) * g[c];
    const float shift = b[c] - m * scale;
    float* ob = out + (long)bi * NNODES * ldout + c;
    for (int n = 0; n < NNODES; ++n)
        ob[(long)n * ldout] = base[(long)n * ldin] * scale + shift;
}

// ---------------------------------------------------------------------------
// Channel-slice copy (for concats not fused into GEMM epilogues)
// ---------------------------------------------------------------------------
__global__ void copy_cols(const float* __restrict__ src, int lds,
                          float* __restrict__ dst, int ldd, int doff,
                          int width, long rows)
{
    long total = rows * (long)width;
    for (long i = (long)blockIdx.x * blockDim.x + threadIdx.x; i < total;
         i += (long)gridDim.x * blockDim.x) {
        long r = i / width;
        int  c = (int)(i - r * width);
        dst[r * ldd + doff + c] = src[r * lds + c];
    }
}

// ---------------------------------------------------------------------------
// Fused tail: (H @ W_ff2 + b_ff2) * W_final, softmax over nodes per batch.
// ---------------------------------------------------------------------------
__global__ void tail_kernel(const float* __restrict__ H,
                            const float* __restrict__ Wff2,
                            const float* __restrict__ bff2,
                            const float* __restrict__ Wfin,
                            float* __restrict__ out)
{
    __shared__ float red[NNODES];
    const int b = blockIdx.x;
    const int n = threadIdx.x;

    const float* h = H + ((long)b * NNODES + n) * 128;
    float acc = 0.0f;
    for (int k = 0; k < 128; ++k) acc += h[k] * Wff2[k];
    const float val = (acc + bff2[0]) * Wfin[0];

    red[n] = val;
    __syncthreads();
    for (int s = NNODES / 2; s > 0; s >>= 1) {
        if (n < s) red[n] = fmaxf(red[n], red[n + s]);
        __syncthreads();
    }
    const float mx = red[0];
    __syncthreads();

    const float e = __expf(val - mx);
    red[n] = e;
    __syncthreads();
    for (int s = NNODES / 2; s > 0; s >>= 1) {
        if (n < s) red[n] += red[n + s];
        __syncthreads();
    }
    out[(long)b * NNODES + n] = e / red[0];
}

// ---------------------------------------------------------------------------
// Host-side launch helpers
// ---------------------------------------------------------------------------
static void gemm(hipStream_t st,
                 const float* A, long lda, long sA,
                 const float* B, long ldb, long sB,
                 float* C, long ldc, long sC, long coff,
                 int M, int K, int Nc, int nb,
                 const float* bias, int relu)
{
    dim3 g((unsigned)(Nc / 64), (unsigned)(M / 64), (unsigned)nb);
    wmma_gemm<<<g, dim3(128), 0, st>>>(A, lda, sA, B, ldb, sB,
                                       C, ldc, sC, coff, M, K, Nc, bias, relu);
}

static void ccopy(hipStream_t st, const float* s, int lds,
                  float* d, int ldd, int doff, int w)
{
    long total = BNROWS * (long)w;
    unsigned blocks = (unsigned)((total + 255) / 256);
    if (blocks > 65535u * 8u) blocks = 65535u * 8u;
    copy_cols<<<blocks, 256, 0, st>>>(s, lds, d, ldd, doff, w, BNROWS);
}

extern "C" void kernel_launch(void* const* d_in, const int* in_sizes, int n_in,
                              void* d_out, int out_size, void* d_ws, size_t ws_size,
                              hipStream_t stream)
{
    (void)in_sizes; (void)n_in; (void)out_size; (void)ws_size;

    const float* x      = (const float*)d_in[0];   // (B,N,1)
    const float* adj    = (const float*)d_in[1];   // (N,N)
    const float* W_fst  = (const float*)d_in[2];   // (1,64)
    const float* W_snd  = (const float*)d_in[3];   // (64,64)
    const float* W_thrd = (const float*)d_in[4];   // (128,128)
    const float* W_thrdb= (const float*)d_in[5];   // (256,256)
    const float* W_frth = (const float*)d_in[6];   // (256,256)
    const float* W_ffth = (const float*)d_in[7];   // (320,256)
    const float* bn6_g  = (const float*)d_in[8];
    const float* bn6_b  = (const float*)d_in[9];
    const float* W_svth = (const float*)d_in[10];  // (256,256)
    const float* bn8_g  = (const float*)d_in[11];
    const float* bn8_b  = (const float*)d_in[12];
    const float* W_nnth = (const float*)d_in[13];  // (512,512)
    const float* W_tnth = (const float*)d_in[14];  // (1024,1024)
    const float* W_lvnth= (const float*)d_in[15];  // (1280,256)
    const float* bn12_g = (const float*)d_in[16];
    const float* bn12_b = (const float*)d_in[17];
    const float* W_ff1  = (const float*)d_in[18];  // (576,128)
    const float* b_ff1  = (const float*)d_in[19];
    const float* W_ff2  = (const float*)d_in[20];  // (128,1)
    const float* b_ff2  = (const float*)d_in[21];
    const float* W_fin  = (const float*)d_in[22];  // (1,1)
    float* out = (float*)d_out;

    // workspace layout (all f32, row-major [B*N, ch])
    float* ws  = (float*)d_ws;
    float* S1  = ws;                   // 64   skip_1
    float* X1  = S1  + 64L   * BNROWS; // 256  A2/A3 growing concat
    float* X2  = X1  + 256L  * BNROWS; // 256  T1 / S2raw
    float* C5  = X2  + 256L  * BNROWS; // 320  [S1 | T2]
    float* S2  = C5  + 320L  * BNROWS; // 256  skip_2 (post-BN6)
    float* A7  = S2  + 256L  * BNROWS; // 512  [S2 | mpl]
    float* A9  = A7  + 512L  * BNROWS; // 1024 [S3a | mpl]
    float* C11 = A9  + 1024L * BNROWS; // 1280 [T3 | S2]
    float* S3r = C11 + 1280L * BNROWS; // 256  pre-BN12
    float* S3  = S3r + 256L  * BNROWS; // 256  skip_3 (post-BN12)
    float* C13 = S3  + 256L  * BNROWS; // 576  [S3 | S2 | S1]
    float* Hb  = C13 + 576L  * BNROWS; // 128  ff1 hidden
    float* Gb  = Hb  + 128L  * BNROWS; // 1024 adjacency agg scratch

    const long M  = BNROWS;
    const long NS = NNODES;

    // 1) skip_1 = x @ W_fst  (outer product: K=1)
    outer64_kernel<<<(unsigned)(BNROWS * 64 / 256), 256, 0, stream>>>(x, W_fst, S1);
    ccopy(stream, S1, 64, X1, 256, 0, 64);

    // 2) green(W_snd): Gb = adj@S1 ; X1[:,64:128] = relu(Gb@W_snd)
    gemm(stream, adj, NNODES, 0, S1, 64, NS * 64, Gb, 64, NS * 64, 0,
         NNODES, NNODES, 64, NBATCH, nullptr, 0);
    gemm(stream, Gb, 64, 0, W_snd, 64, 0, X1, 256, 0, 64, (int)M, 64, 64, 1, nullptr, 1);

    // 3) green(W_thrd): Gb = adj@X1[:,:128] ; X1[:,128:256] = relu(Gb@W_thrd)
    gemm(stream, adj, NNODES, 0, X1, 256, NS * 256, Gb, 128, NS * 128, 0,
         NNODES, NNODES, 128, NBATCH, nullptr, 0);
    gemm(stream, Gb, 128, 0, W_thrd, 128, 0, X1, 256, 0, 128, (int)M, 128, 128, 1, nullptr, 1);

    // 4) T1 = mpl(W_thrdb): Gb = adj@X1 ; X2 = relu(Gb@W_thrdb)
    gemm(stream, adj, NNODES, 0, X1, 256, NS * 256, Gb, 256, NS * 256, 0,
         NNODES, NNODES, 256, NBATCH, nullptr, 0);
    gemm(stream, Gb, 256, 0, W_thrdb, 256, 0, X2, 256, 0, 0, (int)M, 256, 256, 1, nullptr, 1);

    // 5) T2 = mpl(W_frth): Gb = adj@X2 ; C5[:,64:320] = relu(Gb@W_frth)
    gemm(stream, adj, NNODES, 0, X2, 256, NS * 256, Gb, 256, NS * 256, 0,
         NNODES, NNODES, 256, NBATCH, nullptr, 0);
    gemm(stream, Gb, 256, 0, W_frth, 256, 0, C5, 320, 0, 64, (int)M, 256, 256, 1, nullptr, 1);
    ccopy(stream, S1, 64, C5, 320, 0, 64);

    // 6) S2raw = C5 @ W_ffth ; S2 = BN6(S2raw)
    gemm(stream, C5, 320, 0, W_ffth, 256, 0, X2, 256, 0, 0, (int)M, 320, 256, 1, nullptr, 0);
    bn_kernel<<<NBATCH, 256, 0, stream>>>(X2, 256, S2, 256, bn6_g, bn6_b, 256);

    // 7) green(W_svth): A7 = [S2 | relu((adj@S2)@W_svth)]
    ccopy(stream, S2, 256, A7, 512, 0, 256);
    gemm(stream, adj, NNODES, 0, S2, 256, NS * 256, Gb, 256, NS * 256, 0,
         NNODES, NNODES, 256, NBATCH, nullptr, 0);
    gemm(stream, Gb, 256, 0, W_svth, 256, 0, A7, 512, 0, 256, (int)M, 256, 256, 1, nullptr, 1);

    // 8) S3a = BN8(A7) -> A9[:, :512]
    bn_kernel<<<NBATCH, 512, 0, stream>>>(A7, 512, A9, 1024, bn8_g, bn8_b, 512);

    // 9) green(W_nnth): A9[:,512:1024] = relu((adj@S3a)@W_nnth)
    gemm(stream, adj, NNODES, 0, A9, 1024, NS * 1024, Gb, 512, NS * 512, 0,
         NNODES, NNODES, 512, NBATCH, nullptr, 0);
    gemm(stream, Gb, 512, 0, W_nnth, 512, 0, A9, 1024, 0, 512, (int)M, 512, 512, 1, nullptr, 1);

    // 10) T3 = mpl(W_tnth): Gb = adj@A9 ; C11[:, :1024] = relu(Gb@W_tnth)
    gemm(stream, adj, NNODES, 0, A9, 1024, NS * 1024, Gb, 1024, NS * 1024, 0,
         NNODES, NNODES, 1024, NBATCH, nullptr, 0);
    gemm(stream, Gb, 1024, 0, W_tnth, 1024, 0, C11, 1280, 0, 0, (int)M, 1024, 1024, 1, nullptr, 1);
    ccopy(stream, S2, 256, C11, 1280, 1024, 256);

    // 11) S3raw = C11 @ W_lvnth ; S3 = BN12(S3raw)
    gemm(stream, C11, 1280, 0, W_lvnth, 256, 0, S3r, 256, 0, 0, (int)M, 1280, 256, 1, nullptr, 0);
    bn_kernel<<<NBATCH, 256, 0, stream>>>(S3r, 256, S3, 256, bn12_g, bn12_b, 256);

    // 12) C13 = [S3 | S2 | S1]
    ccopy(stream, S3, 256, C13, 576, 0, 256);
    ccopy(stream, S2, 256, C13, 576, 256, 256);
    ccopy(stream, S1, 64, C13, 576, 512, 64);

    // 13) H = relu(C13 @ W_ff1 + b_ff1)
    gemm(stream, C13, 576, 0, W_ff1, 128, 0, Hb, 128, 0, 0, (int)M, 576, 128, 1, b_ff1, 1);

    // 14) tail: (H @ W_ff2 + b_ff2) * W_final -> softmax over nodes
    tail_kernel<<<NBATCH, NNODES, 0, stream>>>(Hb, W_ff2, b_ff2, W_fin, out);
}
